// NormalizedDelinear_39702677684623
// MI455X (gfx1250) — compile-verified, matrix operand order checked by reading, not run on hardware
//
#include <hip/hip_runtime.h>
#include <math.h>

// ---------------------------------------------------------------------------
// Problem constants
// ---------------------------------------------------------------------------
#define NROWS   8192
#define DIM     4096
#define BLK     512
#define XROWS   65536          // NROWS*DIM/BLK
#define WROWS   32768          // 4096*4096/512
#define EPSV    1e-5f
#define KSPLIT  16

typedef __attribute__((ext_vector_type(16))) __bf16 v16bf;
typedef __attribute__((ext_vector_type(8)))  float  v8f;
typedef __attribute__((ext_vector_type(2)))  float  v2f;
typedef __attribute__((ext_vector_type(16))) float  v16f;
typedef __attribute__((ext_vector_type(4)))  int    v4i;

#if defined(__gfx1250__) && __has_builtin(__builtin_amdgcn_wmma_f32_16x16x32_bf16)
#define HAVE_WMMA_BF16 1
#else
#define HAVE_WMMA_BF16 0
#endif

#if defined(__gfx1250__) && __has_builtin(__builtin_amdgcn_wmma_f32_16x16x4_f32)
#define HAVE_WMMA_F32 1
#else
#define HAVE_WMMA_F32 0
#endif

// ---------------------------------------------------------------------------
// CDNA5 async global->LDS copy (ASYNCcnt path) + wait helper.
// Probe-learned builtin signature: (v4i AS(1)* src, v4i AS(3)* dst, imm, imm).
// ---------------------------------------------------------------------------
__device__ __forceinline__ void async_copy_b128(const void* gsrc, void* ldst) {
#if defined(__gfx1250__) && __has_builtin(__builtin_amdgcn_global_load_async_to_lds_b128)
    __builtin_amdgcn_global_load_async_to_lds_b128(
        (__attribute__((address_space(1))) v4i*)gsrc,
        (__attribute__((address_space(3))) v4i*)ldst, 0, 0);
#elif defined(__gfx1250__)
    asm volatile("global_load_async_to_lds_b128 %0, %1, off"
                 :: "v"((unsigned)(size_t)ldst),
                    "v"((unsigned long long)(size_t)gsrc)
                 : "memory");
#else
    __builtin_memcpy(ldst, gsrc, 16);
#endif
}

__device__ __forceinline__ void wait_async0() {
#if defined(__gfx1250__) && __has_builtin(__builtin_amdgcn_s_wait_asynccnt)
    __builtin_amdgcn_s_wait_asynccnt(0);
#elif defined(__gfx1250__)
    asm volatile("s_wait_asynccnt 0" ::: "memory");
#endif
}

// ---------------------------------------------------------------------------
// Utility: zero a float buffer
// ---------------------------------------------------------------------------
__global__ void zero_kernel(float* __restrict__ p, int n) {
    int i = blockIdx.x * blockDim.x + threadIdx.x;
    if (i < n) p[i] = 0.0f;
}

// ---------------------------------------------------------------------------
// 1) LayerNorm (unbiased std + eps): b128 traffic both directions.
//    Thread t handles 16 contiguous elements [t*16, t*16+16).
//    Column of flat element d is d%512 = (t%32)*16 + e  -> LDS ds_add_f32,
//    then one global atomic per column per block.
// ---------------------------------------------------------------------------
__global__ __launch_bounds__(256)
void ln_kernel(const float* __restrict__ x, __bf16* __restrict__ xn_bf,
               float* __restrict__ col_sums) {
    const int row = blockIdx.x;
    const int t   = threadIdx.x;
    const float* xr = x + (size_t)row * DIM + t * 16;

    float v[16];
    float s = 0.f, s2 = 0.f;
#pragma unroll
    for (int q = 0; q < 4; ++q) {
        float4 f = ((const float4*)xr)[q];
        v[4 * q + 0] = f.x; v[4 * q + 1] = f.y;
        v[4 * q + 2] = f.z; v[4 * q + 3] = f.w;
        s  += f.x + f.y + f.z + f.w;
        s2 += f.x * f.x + f.y * f.y + f.z * f.z + f.w * f.w;
    }

    __shared__ float r0[256], r1[256];
    __shared__ float csum[BLK];
    r0[t] = s; r1[t] = s2;
    csum[t] = 0.f; csum[t + 256] = 0.f;
    __syncthreads();
    for (int o = 128; o > 0; o >>= 1) {
        if (t < o) { r0[t] += r0[t + o]; r1[t] += r1[t + o]; }
        __syncthreads();
    }
    const float mean = r0[0] * (1.0f / (float)DIM);
    const float var  = (r1[0] - (float)DIM * mean * mean) * (1.0f / (float)(DIM - 1));
    const float inv  = 1.0f / (sqrtf(var) + EPSV);

    const int cbase = (t & 31) * 16;
    v16bf o;
#pragma unroll
    for (int e = 0; e < 16; ++e) {
        float y = (v[e] - mean) * inv;
        o[e] = (__bf16)y;
        atomicAdd(&csum[cbase + e], y);          // ds_add_f32
    }
    *(v16bf*)(xn_bf + (size_t)row * DIM + t * 16) = o;   // 32B store

    __syncthreads();
    atomicAdd(&col_sums[t],       csum[t]);
    atomicAdd(&col_sums[t + 256], csum[t + 256]);
}

// ---------------------------------------------------------------------------
// 2a) X_mean = col_sums / XROWS
// ---------------------------------------------------------------------------
__global__ void xmean_kernel(const float* __restrict__ col_sums,
                             float* __restrict__ Xm) {
    int t = blockIdx.x * blockDim.x + threadIdx.x;
    if (t < BLK) Xm[t] = col_sums[t] * (1.0f / (float)XROWS);
}

// ---------------------------------------------------------------------------
// 2b) G += X^T X over a K-slice (split-K, bf16 WMMA, f32 atomics).
//     K-contiguity requires a transpose; X stays L2-resident (64 MB bf16),
//     strided lane loads acceptable for this 34 GFLOP term.
// ---------------------------------------------------------------------------
__global__ __launch_bounds__(32)
void xtx_kernel(const __bf16* __restrict__ X, float* __restrict__ G) {
    const int lane  = threadIdx.x & 31;
    const int i0    = blockIdx.y * 16;
    const int j0    = blockIdx.x * 16;
    const int kchu  = XROWS / KSPLIT;
    const int kbeg  = blockIdx.z * kchu;
    const int m     = lane & 15;
    const int hi    = lane >> 4;
    const int khalf = hi * 16;

    v8f acc = {};
#if HAVE_WMMA_BF16
    for (int k = kbeg; k < kbeg + kchu; k += 32) {
        v16bf a, b;
#pragma unroll
        for (int e = 0; e < 16; ++e) {
            const size_t kk = (size_t)(k + khalf + e) * BLK;
            a[e] = X[kk + i0 + m];
            b[e] = X[kk + j0 + m];
        }
        acc = __builtin_amdgcn_wmma_f32_16x16x32_bf16(false, a, false, b,
                                                      (short)0, acc, false, false);
    }
#else
#pragma unroll
    for (int r = 0; r < 8; ++r) {
        float t2 = 0.f;
        const int row = i0 + r + 8 * hi, col = j0 + m;
        for (int k = kbeg; k < kbeg + kchu; ++k)
            t2 += (float)X[(size_t)k * BLK + row] * (float)X[(size_t)k * BLK + col];
        acc[r] = t2;
    }
#endif
    const int col = j0 + m;
#pragma unroll
    for (int r = 0; r < 8; ++r) {
        const int row = i0 + r + 8 * hi;
        atomicAdd(&G[(size_t)row * BLK + col], acc[r]);
    }
}

// ---------------------------------------------------------------------------
// 2c) cov = eps*I + G/n - Xm Xm^T ; accumulate ||cov||_F^2
// ---------------------------------------------------------------------------
__global__ __launch_bounds__(256)
void cov_kernel(const float* __restrict__ G, const float* __restrict__ Xm,
                float* __restrict__ cov, float* __restrict__ sumsq) {
    const int i = blockIdx.x;
    const int t = threadIdx.x;
    const float xi = Xm[i];
    float local = 0.f;
#pragma unroll
    for (int jj = 0; jj < 2; ++jj) {
        int j = t + 256 * jj;
        float c = G[(size_t)i * BLK + j] * (1.0f / (float)XROWS) - xi * Xm[j];
        if (i == j) c += EPSV;
        cov[(size_t)i * BLK + j] = c;
        local += c * c;
    }
    __shared__ float red[256];
    red[t] = local;
    __syncthreads();
    for (int o = 128; o > 0; o >>= 1) {
        if (t < o) red[t] += red[t + o];
        __syncthreads();
    }
    if (t == 0) atomicAdd(sumsq, red[0]);
}

// ---------------------------------------------------------------------------
// 3a) Y0 = cov/||cov||_F ; Z0 = I
// ---------------------------------------------------------------------------
__global__ void initYZ_kernel(const float* __restrict__ cov,
                              const float* __restrict__ sumsq,
                              float* __restrict__ Y, float* __restrict__ Z) {
    int i = blockIdx.x * blockDim.x + threadIdx.x;
    if (i >= BLK * BLK) return;
    const float inv = 1.0f / sqrtf(sumsq[0]);
    Y[i] = cov[i] * inv;
    Z[i] = ((i / BLK) == (i % BLK)) ? 1.0f : 0.0f;
}

// ---------------------------------------------------------------------------
// 3b) 512x512x512 fp32 GEMM with V_WMMA_F32_16X16X4_F32 (keeps the
//     numerically-sensitive Denman-Beavers iteration in fp32; only ~4 GFLOP).
//     mode 0: C = A@B ;  mode 1: C = 1.5*I - 0.5*(A@B)
// ---------------------------------------------------------------------------
__global__ __launch_bounds__(256)
void ns_gemm_kernel(const float* __restrict__ A, const float* __restrict__ B,
                    float* __restrict__ C, int mode) {
    const int wave = threadIdx.x >> 5;
    const int lane = threadIdx.x & 31;
    const int tile = blockIdx.x * 8 + wave;       // 0..1023
    const int i0   = (tile >> 5) * 16;
    const int j0   = (tile & 31) * 16;
    const int m    = lane & 15;
    const int hi   = lane >> 4;

    v8f acc = {};
#if HAVE_WMMA_F32
    for (int k = 0; k < BLK; k += 4) {
        v2f a, b;
        a[0] = A[(size_t)(i0 + m) * BLK + k + 2 * hi];
        a[1] = A[(size_t)(i0 + m) * BLK + k + 2 * hi + 1];
        b[0] = B[(size_t)(k + 2 * hi) * BLK + j0 + m];
        b[1] = B[(size_t)(k + 2 * hi + 1) * BLK + j0 + m];
        acc = __builtin_amdgcn_wmma_f32_16x16x4_f32(false, a, false, b,
                                                    (short)0, acc, false, false);
    }
#else
#pragma unroll
    for (int r = 0; r < 8; ++r) {
        float t2 = 0.f;
        const int row = i0 + r + 8 * hi, col = j0 + m;
        for (int k = 0; k < BLK; ++k)
            t2 += A[(size_t)row * BLK + k] * B[(size_t)k * BLK + col];
        acc[r] = t2;
    }
#endif
    const int col = j0 + m;
#pragma unroll
    for (int r = 0; r < 8; ++r) {
        const int row = i0 + r + 8 * hi;
        float v = acc[r];
        if (mode == 1) v = -0.5f * v + ((row == col) ? 1.5f : 0.0f);
        C[(size_t)row * BLK + col] = v;
    }
}

// ---------------------------------------------------------------------------
// 3c) S = Z * ||cov||_F^{-1/2}  -> bf16
// ---------------------------------------------------------------------------
__global__ void isqrt_scale_kernel(const float* __restrict__ Z,
                                   const float* __restrict__ sumsq,
                                   __bf16* __restrict__ Sb) {
    int i = blockIdx.x * blockDim.x + threadIdx.x;
    if (i >= BLK * BLK) return;
    const float scale = 1.0f / sqrtf(sqrtf(sumsq[0]));  // (||A||^2)^(-1/4)
    Sb[i] = (__bf16)(Z[i] * scale);
}

// ---------------------------------------------------------------------------
// 4) w = weight.reshape(32768,512) @ S   (bf16 WMMA; A via 64B vector loads;
//    S is 512KB -> L2-resident, strided lane loads fine)
// ---------------------------------------------------------------------------
__global__ __launch_bounds__(256)
void wtrans_kernel(const float* __restrict__ weight, const __bf16* __restrict__ S,
                   float* __restrict__ w_f32, __bf16* __restrict__ w_bf) {
    const int wave  = threadIdx.x >> 5;
    const int lane  = threadIdx.x & 31;
    const int i0    = (blockIdx.y * 8 + wave) * 16;    // gridDim.y = 256
    const int j0    = blockIdx.x * 16;                 // gridDim.x = 32
    const int m     = lane & 15;
    const int hi    = lane >> 4;
    const int khalf = hi * 16;

    v8f acc = {};
#if HAVE_WMMA_BF16
    const float* wrow = weight + (size_t)(i0 + m) * BLK;
    for (int k = 0; k < BLK; k += 32) {
        v16f af = *(const v16f*)(wrow + k + khalf);     // 64B vector load
        v16bf a, b;
#pragma unroll
        for (int e = 0; e < 16; ++e) {
            a[e] = (__bf16)af[e];
            b[e] = S[(size_t)(k + khalf + e) * BLK + j0 + m];
        }
        acc = __builtin_amdgcn_wmma_f32_16x16x32_bf16(false, a, false, b,
                                                      (short)0, acc, false, false);
    }
#else
#pragma unroll
    for (int r = 0; r < 8; ++r) {
        float t2 = 0.f;
        const int row = i0 + r + 8 * hi, col = j0 + m;
        for (int k = 0; k < BLK; ++k)
            t2 += weight[(size_t)row * BLK + k] * (float)S[(size_t)k * BLK + col];
        acc[r] = t2;
    }
#endif
    const int col = j0 + m;
#pragma unroll
    for (int r = 0; r < 8; ++r) {
        const size_t flat = (size_t)(i0 + r + 8 * hi) * BLK + col;
        w_f32[flat] = acc[r];
        w_bf[flat]  = (__bf16)acc[r];
    }
}

// ---------------------------------------------------------------------------
// 5) b[o] = bias[o] - sum_d w[o,d] * Xm[d % 512]
// ---------------------------------------------------------------------------
__global__ __launch_bounds__(256)
void bias_kernel(const float* __restrict__ w_f32, const float* __restrict__ Xm,
                 const float* __restrict__ bias, float* __restrict__ bvec) {
    const int o = blockIdx.x;
    const int t = threadIdx.x;
    float s = 0.f;
#pragma unroll
    for (int i = 0; i < 16; ++i) {
        int d = t + 256 * i;
        s += w_f32[(size_t)o * DIM + d] * Xm[d & (BLK - 1)];
    }
    __shared__ float red[256];
    red[t] = s;
    __syncthreads();
    for (int off = 128; off > 0; off >>= 1) {
        if (t < off) red[t] += red[t + off];
        __syncthreads();
    }
    if (t == 0) bvec[o] = bias[o] - red[0];
}

// ---------------------------------------------------------------------------
// 6) out = xn @ w^T + b   (M=8192, N=4096, K=4096, bf16 WMMA)
//    Block = 8 waves sharing one N-tile (j0). The 16xKC B strip is staged in
//    LDS once per block via GLOBAL_LOAD_ASYNC_TO_LDS_B128 (ASYNCcnt), then
//    each wave reads conflict-lite 32B ds fragments. A fragments are
//    contiguous 32B global vector loads + prefetch.
//    LDS row stride 80 bf16 (160B): 32B-aligned fragments, ~2-way conflicts.
// ---------------------------------------------------------------------------
#define KC 64
#define BSTRIDE 80

__global__ __launch_bounds__(256)
void out_gemm_kernel(const __bf16* __restrict__ Xn, const __bf16* __restrict__ Wb,
                     const float* __restrict__ bvec, float* __restrict__ out) {
    __shared__ __attribute__((aligned(32))) __bf16 Btile[16 * BSTRIDE];

    const int t     = threadIdx.x;
    const int wave  = t >> 5;
    const int lane  = t & 31;
    const int i0    = (blockIdx.y * 8 + wave) * 16;    // gridDim.y = 64
    const int j0    = blockIdx.x * 16;                 // gridDim.x = 256
    const int m     = lane & 15;
    const int hi    = lane >> 4;
    const int khalf = hi * 16;

    v8f acc = {};
#if HAVE_WMMA_BF16
    const __bf16* arow = Xn + (size_t)(i0 + m) * DIM;
    for (int k = 0; k < DIM; k += KC) {
        __syncthreads();                       // previous reads done
        if (t < 128) {                         // 16 rows x 8 b128 segments
            const int row = t >> 3, seg = t & 7;
            async_copy_b128(Wb + (size_t)(j0 + row) * DIM + k + seg * 8,
                            Btile + row * BSTRIDE + seg * 8);
        }
        wait_async0();                         // s_wait_asynccnt 0
        __syncthreads();                       // all waves' copies visible

        __builtin_prefetch(arow + k + 512, 0, 0);   // global_prefetch_b8
#pragma unroll
        for (int kk = 0; kk < KC; kk += 32) {
            v16bf a = *(const v16bf*)(arow + k + kk + khalf);
            v16bf b = *(const v16bf*)(Btile + m * BSTRIDE + kk + khalf);
            acc = __builtin_amdgcn_wmma_f32_16x16x32_bf16(false, a, false, b,
                                                          (short)0, acc, false, false);
        }
    }
#else
#pragma unroll
    for (int r = 0; r < 8; ++r) {
        float t2 = 0.f;
        const int row = i0 + r + 8 * hi, col = j0 + m;
        for (int k = 0; k < DIM; ++k)
            t2 += (float)Xn[(size_t)row * DIM + k] * (float)Wb[(size_t)col * DIM + k];
        acc[r] = t2;
    }
#endif
    const int   col  = j0 + m;
    const float badd = bvec[col];
#pragma unroll
    for (int r = 0; r < 8; ++r) {
        const int row = i0 + r + 8 * hi;
        out[(size_t)row * DIM + col] = acc[r] + badd;
    }
}

// ---------------------------------------------------------------------------
// Host launcher
// ---------------------------------------------------------------------------
extern "C" void kernel_launch(void* const* d_in, const int* in_sizes, int n_in,
                              void* d_out, int out_size, void* d_ws, size_t ws_size,
                              hipStream_t stream) {
    const float* x      = (const float*)d_in[0];   // [8192, 4096]
    const float* weight = (const float*)d_in[1];   // [4096, 4096]
    const float* bias   = (const float*)d_in[2];   // [4096]
    float* out          = (float*)d_out;           // [8192, 4096]

    char* p = (char*)d_ws;
    auto alloc = [&](size_t bytes) -> void* {
        void* r = (void*)p;
        p += (bytes + 255) & ~(size_t)255;
        return r;
    };
    __bf16* xn_bf   = (__bf16*)alloc((size_t)NROWS * DIM * 2);   // 64 MB
    __bf16* w_bf    = (__bf16*)alloc((size_t)DIM * DIM * 2);     // 32 MB
    float*  w_f32   = (float*) alloc((size_t)DIM * DIM * 4);     // 64 MB
    float*  G       = (float*) alloc((size_t)BLK * BLK * 4);
    float*  cov     = (float*) alloc((size_t)BLK * BLK * 4);
    float*  Y       = (float*) alloc((size_t)BLK * BLK * 4);
    float*  Z       = (float*) alloc((size_t)BLK * BLK * 4);
    float*  T       = (float*) alloc((size_t)BLK * BLK * 4);
    float*  Y2      = (float*) alloc((size_t)BLK * BLK * 4);
    float*  Z2      = (float*) alloc((size_t)BLK * BLK * 4);
    __bf16* Sb      = (__bf16*)alloc((size_t)BLK * BLK * 2);
    float*  colsum  = (float*) alloc(BLK * 4);
    float*  Xm      = (float*) alloc(BLK * 4);
    float*  sumsq   = (float*) alloc(256);
    float*  bvec    = (float*) alloc(DIM * 4);
    (void)in_sizes; (void)n_in; (void)out_size; (void)ws_size;

    // zero the atomic accumulators (fresh every call -> deterministic)
    zero_kernel<<<(BLK * BLK + 255) / 256, 256, 0, stream>>>(G, BLK * BLK);
    zero_kernel<<<2, 256, 0, stream>>>(colsum, BLK);
    zero_kernel<<<1, 256, 0, stream>>>(sumsq, 1);

    // 1) layernorm + bf16 cast + column sums
    ln_kernel<<<NROWS, 256, 0, stream>>>(x, xn_bf, colsum);
    xmean_kernel<<<2, 256, 0, stream>>>(colsum, Xm);

    // 2) G = X^T X (split-K), then cov + Frobenius norm
    xtx_kernel<<<dim3(BLK / 16, BLK / 16, KSPLIT), 32, 0, stream>>>(xn_bf, G);
    cov_kernel<<<BLK, 256, 0, stream>>>(G, Xm, cov, sumsq);

    // 3) Denman-Beavers: Y=A/||A||, Z=I ; 5 iterations of 3 fp32 GEMMs
    initYZ_kernel<<<(BLK * BLK + 255) / 256, 256, 0, stream>>>(cov, sumsq, Y, Z);
    for (int it = 0; it < 5; ++it) {
        ns_gemm_kernel<<<128, 256, 0, stream>>>(Z, Y, T, 1);   // T = 1.5I - 0.5 Z@Y
        ns_gemm_kernel<<<128, 256, 0, stream>>>(Y, T, Y2, 0);  // Y' = Y@T
        ns_gemm_kernel<<<128, 256, 0, stream>>>(T, Z, Z2, 0);  // Z' = T@Z
        float* tmp;
        tmp = Y; Y = Y2; Y2 = tmp;
        tmp = Z; Z = Z2; Z2 = tmp;
    }
    isqrt_scale_kernel<<<(BLK * BLK + 255) / 256, 256, 0, stream>>>(Z, sumsq, Sb);

    // 4) w = weight.reshape(32768,512) @ S
    wtrans_kernel<<<dim3(BLK / 16, WROWS / 16 / 8), 256, 0, stream>>>(weight, Sb, w_f32, w_bf);

    // 5) b = bias - (w @ Xm-tiled row-sum)
    bias_kernel<<<DIM, 256, 0, stream>>>(w_f32, Xm, bias, bvec);

    // 6) out = xn @ w^T + b (async-LDS staged B)
    out_gemm_kernel<<<dim3(DIM / 16, NROWS / 16 / 8), 256, 0, stream>>>(xn_bf, w_bf, bvec, out);
}
